// CGBlock_20469814133507
// MI455X (gfx1250) — compile-verified, matrix operand order checked by reading, not run on hardware
//
#include <hip/hip_runtime.h>

// ---------------------------------------------------------------------------
// CGBlock: s_mlp/c_mlp (Linear->BN->ReLU x2 -> Linear), context gate, max-pool
// B=256, S=512, D=128.  bf16 WMMA (v_wmma_f32_16x16x32_bf16), f32 accumulate.
// One workgroup (512 threads = 16 wave32) per sequence slice.
// Weights double-buffered in LDS and moved with global_load_async_to_lds_b128
// (ASYNCcnt) so all weight staging overlaps WMMA compute. ~141KB LDS per WG.
// ---------------------------------------------------------------------------

typedef __attribute__((ext_vector_type(16))) __bf16 v16bf;
typedef __attribute__((ext_vector_type(8)))  __bf16 v8bf;
typedef __attribute__((ext_vector_type(4)))  __bf16 v4bf;
typedef __attribute__((ext_vector_type(8)))  float  v8f;
typedef __attribute__((ext_vector_type(4)))  float  v4f;

#define BATCH    256
#define DIM      128
#define SEQ      512
#define LDX      136          // padded LDS row stride (bf16 elems) = 272B
#define NTILE    8            // N tiles (8*16 = 128) per wave
#define KSTEPS   4            // 128 / 32
#define NTHREADS 512          // 16 wave32, one M-tile each
#define BN_EPS   1e-5f

#define XB_BYTES (BATCH * LDX * 2)
#define WB_BYTES (DIM * LDX * 2)
#define SMEM_BYTES (XB_BYTES + 2 * WB_BYTES + 4 * DIM * 4)

// ---- CDNA5 async global->LDS copy (ASYNCcnt) --------------------------------
// Dynamic LDS starts at offset 0 of the wave's LDS allocation, so byte offsets
// within smem are valid VDST lds addresses.
__device__ __forceinline__ void async_b128_to_lds(unsigned lds_off,
                                                  const void* gaddr) {
    asm volatile("global_load_async_to_lds_b128 %0, %1, off"
                 :: "v"(lds_off), "v"((unsigned long long)gaddr)
                 : "memory");
}
template <int N>
__device__ __forceinline__ void wait_asynccnt() {
    asm volatile("s_wait_asynccnt %0" :: "i"(N) : "memory");
}

// Issue the async copy of one 128x128 bf16 weight matrix into an LDS tile
// (row stride LDX).  4 wave-instructions per wave; lanes carry consecutive
// 16B chunks -> 512B contiguous global per instruction.
__device__ __forceinline__ void async_stage_w(unsigned wb_off,
                                              const __bf16* __restrict__ wsrc,
                                              int tid) {
    #pragma unroll
    for (int i = 0; i < 4; ++i) {
        int c = tid + i * NTHREADS;                         // chunk of 8 bf16
        unsigned lds = wb_off + (unsigned)(((c >> 4) * LDX + (c & 15) * 8) * 2);
        async_b128_to_lds(lds, (const void*)(wsrc + c * 8));
    }
}

// Build a 16x32 bf16 WMMA fragment (A or B operand) from an LDS tile stored
// row-major with stride LDX.  Lane mapping per CDNA5 ISA 7.12.2:
//   halves 0-7  <- K = kg .. kg+7   (kg = ks*32 + 8*laneHi)
//   halves 8-15 <- K = kg+16 .. kg+23
__device__ __forceinline__ v16bf load_frag(const __bf16* __restrict__ base,
                                           int row, int kg) {
    v8bf lo = *(const v8bf*)(base + row * LDX + kg);
    v8bf hi = *(const v8bf*)(base + row * LDX + kg + 16);
    return __builtin_shufflevector(lo, hi,
                                   0,1,2,3,4,5,6,7,8,9,10,11,12,13,14,15);
}

template <bool GATE>
__global__ void __launch_bounds__(NTHREADS)
mlp_bn_gate_kernel(const float* __restrict__ xin, int xin_row_stride,
                   const __bf16* __restrict__ wpack,          // 3 x 128x128 bf16
                   const float* __restrict__ b1, const float* __restrict__ g1,
                   const float* __restrict__ be1,
                   const float* __restrict__ b2, const float* __restrict__ g2,
                   const float* __restrict__ be2,
                   const float* __restrict__ b3,
                   const float* __restrict__ gate,            // c_out (GATE only)
                   float* __restrict__ out0,                  // GATE: s_out ; else c_out
                   float* __restrict__ out_agg)               // GATE only
{
    extern __shared__ __align__(16) char smem[];
    __bf16* xb  = (__bf16*)smem;                       // [256][LDX] activations
    __bf16* wb0 = (__bf16*)(smem + XB_BYTES);          // weight buffer A
    __bf16* wb1 = (__bf16*)(smem + XB_BYTES + WB_BYTES); // weight buffer B
    float*  st  = (float*)(smem + XB_BYTES + 2 * WB_BYTES); // [4][128] BN stats

    const int tid   = threadIdx.x;
    const int lane  = tid & 31;
    const int wave  = tid >> 5;                 // 0..15 == M-tile
    const int lm    = lane & 15;
    const int lh    = lane >> 4;
    const int slice = GATE ? blockIdx.x : 0;
    const int arow  = wave * 16 + lm;           // A-operand row for this lane

    // ---- Phase 0: async-stage W1->wb0, W2->wb1; stage x; zero BN stats ----
    async_stage_w(XB_BYTES, wpack, tid);                       // W1 -> wb0
    async_stage_w(XB_BYTES + WB_BYTES, wpack + DIM * DIM, tid); // W2 -> wb1
    for (int c = tid; c < (BATCH * DIM) / 4; c += NTHREADS) {  // 16 iters
        int m = c >> 5, k = (c & 31) * 4;
        v4f x4 = *(const v4f*)(xin + m * xin_row_stride + slice * DIM + k);
        v4bf b4;
        #pragma unroll
        for (int j = 0; j < 4; ++j) b4[j] = (__bf16)x4[j];
        *(v4bf*)(xb + m * LDX + k) = b4;
    }
    if (tid < 4 * DIM) st[tid] = 0.0f;
    wait_asynccnt<4>();          // first 4 copies (W1) landed; W2 still in flight
    __syncthreads();

    v8f acc[NTILE];

    // ================= Layer 1: GEMM + bias (wb0) =================
    #pragma unroll
    for (int nt = 0; nt < NTILE; ++nt) {
        float bv = b1[nt * 16 + lm];
        #pragma unroll
        for (int r = 0; r < 8; ++r) acc[nt][r] = bv;
    }
    #pragma unroll
    for (int ks = 0; ks < KSTEPS; ++ks) {
        int kg = ks * 32 + 8 * lh;
        v16bf af = load_frag(xb, arow, kg);
        #pragma unroll
        for (int nt = 0; nt < NTILE; ++nt) {
            v16bf bf = load_frag(wb0, nt * 16 + lm, kg);
            acc[nt] = __builtin_amdgcn_wmma_f32_16x16x32_bf16(
                false, af, false, bf, (short)0, acc[nt], false, false);
        }
    }
    __syncthreads();                                   // wb0 free now

    // ---- W3 -> wb0 (flies during BN1), BN1 partial sums (ds_add_f32) ----
    async_stage_w(XB_BYTES, wpack + 2 * DIM * DIM, tid);
    #pragma unroll
    for (int nt = 0; nt < NTILE; ++nt) {
        float s = 0.f, s2 = 0.f;
        #pragma unroll
        for (int r = 0; r < 8; ++r) { float v = acc[nt][r]; s += v; s2 += v * v; }
        atomicAdd(&st[nt * 16 + lm], s);
        atomicAdd(&st[DIM + nt * 16 + lm], s2);
    }
    __syncthreads();

    // ---- BN1 normalize + ReLU -> xb (bf16) ----
    #pragma unroll
    for (int nt = 0; nt < NTILE; ++nt) {
        int n = nt * 16 + lm;
        float mean = st[n] * (1.0f / BATCH);
        float var  = st[DIM + n] * (1.0f / BATCH) - mean * mean;
        float sc = g1[n] * rsqrtf(var + BN_EPS);
        float sh = be1[n] - mean * sc;
        #pragma unroll
        for (int r = 0; r < 8; ++r) {
            float v = acc[nt][r] * sc + sh;
            v = v > 0.f ? v : 0.f;
            xb[(wave * 16 + r + 8 * lh) * LDX + n] = (__bf16)v;
        }
    }
    wait_asynccnt<4>();          // W2 (issued before W3) fully landed
    __syncthreads();

    // ================= Layer 2: GEMM + bias (wb1) =================
    #pragma unroll
    for (int nt = 0; nt < NTILE; ++nt) {
        float bv = b2[nt * 16 + lm];
        #pragma unroll
        for (int r = 0; r < 8; ++r) acc[nt][r] = bv;
    }
    #pragma unroll
    for (int ks = 0; ks < KSTEPS; ++ks) {
        int kg = ks * 32 + 8 * lh;
        v16bf af = load_frag(xb, arow, kg);
        #pragma unroll
        for (int nt = 0; nt < NTILE; ++nt) {
            v16bf bf = load_frag(wb1, nt * 16 + lm, kg);
            acc[nt] = __builtin_amdgcn_wmma_f32_16x16x32_bf16(
                false, af, false, bf, (short)0, acc[nt], false, false);
        }
    }
    __syncthreads();

    // ---- BN2 partial sums ----
    #pragma unroll
    for (int nt = 0; nt < NTILE; ++nt) {
        float s = 0.f, s2 = 0.f;
        #pragma unroll
        for (int r = 0; r < 8; ++r) { float v = acc[nt][r]; s += v; s2 += v * v; }
        atomicAdd(&st[2 * DIM + nt * 16 + lm], s);
        atomicAdd(&st[3 * DIM + nt * 16 + lm], s2);
    }
    __syncthreads();

    // ---- BN2 normalize + ReLU -> xb ----
    #pragma unroll
    for (int nt = 0; nt < NTILE; ++nt) {
        int n = nt * 16 + lm;
        float mean = st[2 * DIM + n] * (1.0f / BATCH);
        float var  = st[3 * DIM + n] * (1.0f / BATCH) - mean * mean;
        float sc = g2[n] * rsqrtf(var + BN_EPS);
        float sh = be2[n] - mean * sc;
        #pragma unroll
        for (int r = 0; r < 8; ++r) {
            float v = acc[nt][r] * sc + sh;
            v = v > 0.f ? v : 0.f;
            xb[(wave * 16 + r + 8 * lh) * LDX + n] = (__bf16)v;
        }
    }
    wait_asynccnt<0>();          // W3 landed
    __syncthreads();

    // ================= Layer 3: GEMM + bias (wb0, no BN) =================
    #pragma unroll
    for (int nt = 0; nt < NTILE; ++nt) {
        float bv = b3[nt * 16 + lm];
        #pragma unroll
        for (int r = 0; r < 8; ++r) acc[nt][r] = bv;
    }
    #pragma unroll
    for (int ks = 0; ks < KSTEPS; ++ks) {
        int kg = ks * 32 + 8 * lh;
        v16bf af = load_frag(xb, arow, kg);
        #pragma unroll
        for (int nt = 0; nt < NTILE; ++nt) {
            v16bf bf = load_frag(wb0, nt * 16 + lm, kg);
            acc[nt] = __builtin_amdgcn_wmma_f32_16x16x32_bf16(
                false, af, false, bf, (short)0, acc[nt], false, false);
        }
    }

    // ---- Epilogue: gate, store, max-pool ----
    #pragma unroll
    for (int nt = 0; nt < NTILE; ++nt) {
        int n = nt * 16 + lm;
        #pragma unroll
        for (int r = 0; r < 8; ++r) {
            int row = wave * 16 + r + 8 * lh;   // batch index
            float v = acc[nt][r];
            if (GATE) {
                v *= gate[row * DIM + n];
                out0[row * (SEQ * DIM) + slice * DIM + n] = v;
                atomicMax(&out_agg[row * DIM + n], v);   // seq max-pool
            } else {
                out0[row * DIM + n] = v;                 // c_out to workspace
            }
        }
    }
}

// Convert all six 128x128 weight matrices f32 -> bf16 once (vectorized x4).
__global__ void cvt_weights_kernel(const float* __restrict__ w0, const float* __restrict__ w1,
                                   const float* __restrict__ w2, const float* __restrict__ w3,
                                   const float* __restrict__ w4, const float* __restrict__ w5,
                                   __bf16* __restrict__ dst) {
    int i = blockIdx.x * blockDim.x + threadIdx.x;
    if (i >= (6 * DIM * DIM) / 4) return;
    const float* srcs[6] = {w0, w1, w2, w3, w4, w5};
    v4f x4 = *(const v4f*)(srcs[i >> 12] + (i & 4095) * 4);
    v4bf b4;
    #pragma unroll
    for (int j = 0; j < 4; ++j) b4[j] = (__bf16)x4[j];
    ((v4bf*)dst)[i] = b4;
}

__global__ void init_agg_kernel(float* __restrict__ agg) {
    int i = blockIdx.x * blockDim.x + threadIdx.x;
    if (i < BATCH * DIM) agg[i] = -3.402823466e38f;
}

extern "C" void kernel_launch(void* const* d_in, const int* in_sizes, int n_in,
                              void* d_out, int out_size, void* d_ws, size_t ws_size,
                              hipStream_t stream) {
    (void)in_sizes; (void)n_in; (void)out_size; (void)ws_size;

    const float* s     = (const float*)d_in[0];
    const float* c     = (const float*)d_in[1];
    const float* s_w1  = (const float*)d_in[2];
    const float* s_b1  = (const float*)d_in[3];
    const float* s_g1  = (const float*)d_in[4];
    const float* s_be1 = (const float*)d_in[5];
    const float* s_w2  = (const float*)d_in[6];
    const float* s_b2  = (const float*)d_in[7];
    const float* s_g2  = (const float*)d_in[8];
    const float* s_be2 = (const float*)d_in[9];
    const float* s_w3  = (const float*)d_in[10];
    const float* s_b3  = (const float*)d_in[11];
    const float* c_w1  = (const float*)d_in[12];
    const float* c_b1  = (const float*)d_in[13];
    const float* c_g1  = (const float*)d_in[14];
    const float* c_be1 = (const float*)d_in[15];
    const float* c_w2  = (const float*)d_in[16];
    const float* c_b2  = (const float*)d_in[17];
    const float* c_g2  = (const float*)d_in[18];
    const float* c_be2 = (const float*)d_in[19];
    const float* c_w3  = (const float*)d_in[20];
    const float* c_b3  = (const float*)d_in[21];

    float* out_s   = (float*)d_out;                       // [256,512,128]
    float* out_agg = out_s + (size_t)BATCH * SEQ * DIM;   // [256,128]

    float*  ws_cout = (float*)d_ws;                                   // 128 KB
    __bf16* wpack   = (__bf16*)((char*)d_ws + BATCH * DIM * sizeof(float));
    // wpack: [s_w1 | s_w2 | s_w3 | c_w1 | c_w2 | c_w3], each 128x128 bf16

    // allow >64KB dynamic LDS (gfx1250 WGP has 320KB)
    hipFuncSetAttribute((const void*)mlp_bn_gate_kernel<false>,
                        hipFuncAttributeMaxDynamicSharedMemorySize, SMEM_BYTES);
    hipFuncSetAttribute((const void*)mlp_bn_gate_kernel<true>,
                        hipFuncAttributeMaxDynamicSharedMemorySize, SMEM_BYTES);

    cvt_weights_kernel<<<((6 * DIM * DIM) / 4 + 255) / 256, 256, 0, stream>>>(
        s_w1, s_w2, s_w3, c_w1, c_w2, c_w3, wpack);
    init_agg_kernel<<<(BATCH * DIM + 255) / 256, 256, 0, stream>>>(out_agg);

    // c_mlp: one [256,128] "slice", un-gated, -> workspace
    mlp_bn_gate_kernel<false><<<1, NTHREADS, SMEM_BYTES, stream>>>(
        c, DIM, wpack + 3 * DIM * DIM,
        c_b1, c_g1, c_be1, c_b2, c_g2, c_be2, c_b3,
        nullptr, ws_cout, nullptr);

    // s_mlp: 512 slices, gated by c_out, max-pooled into out_agg
    mlp_bn_gate_kernel<true><<<SEQ, NTHREADS, SMEM_BYTES, stream>>>(
        s, SEQ * DIM, wpack,
        s_b1, s_g1, s_be1, s_b2, s_g2, s_be2, s_b3,
        ws_cout, out_s, out_agg);
}
